// MambaSSM_90194313216009
// MI455X (gfx1250) — matrix operand order repaired and checked by reading, not compiled, not run on hardware
//
#include <hip/hip_runtime.h>

#define D_MODEL 1024
#define D_STATE 16
#define D_CONV  4
#define D_INNER 2048
#define TSEQ    2048
#define NPROJ      (D_INNER + 2*D_STATE)   // 2080 (true columns)
#define NPROJ_PAD  2112                    // 33 * 64, branch-free GEMM tiles
#define NCHUNK  32
#define CLEN    64                         // NCHUNK*CLEN == TSEQ
#define KCH     128                        // K-chunk staged by TDM
#define TILE_BYTES (16 * KCH * 2)          // 4096 B per staged A tile

typedef __bf16 v16bf __attribute__((ext_vector_type(16)));
typedef float  v8f   __attribute__((ext_vector_type(8)));
typedef unsigned int u32;
typedef u32 u32x4 __attribute__((ext_vector_type(4)));
typedef u32 u32x8 __attribute__((ext_vector_type(8)));

// ---------------- helpers ----------------
__device__ __forceinline__ unsigned short f2bf(float f) {
    u32 u = __float_as_uint(f);
    return (unsigned short)((u + 0x7FFFu + ((u >> 16) & 1u)) >> 16);  // RNE
}
__device__ __forceinline__ float sigm(float x)  { return 1.f / (1.f + __expf(-x)); }
__device__ __forceinline__ float softp(float x) { return (x > 20.f) ? x : log1pf(__expf(x)); }

// Tensor Data Mover: 2D tile (rows x row_elems, bf16) global -> LDS.
// D# group0/group1 packed per CDNA5 ISA 8.3/8.4. VADDR2/3 disabled (<=2D).
__device__ __forceinline__ void tdm_load_tile_2d(u32 lds_byte, const void* gsrc,
                                                 u32 row_elems, u32 rows,
                                                 u32 tensor_d0, u32 tensor_d1,
                                                 unsigned long long stride_elems) {
    unsigned long long ga = (unsigned long long)gsrc;
    u32x4 g0;
    g0[0] = 1u;                                              // count=1, user D#
    g0[1] = lds_byte;                                        // lds_addr (bytes)
    g0[2] = (u32)ga;                                         // global_addr[31:0]
    g0[3] = (u32)((ga >> 32) & 0x01FFFFFFu) | (2u << 30);    // addr[56:32] | type=2
    u32x8 g1;
    g1[0] = 1u << 16;                                        // wg_mask=0, data_size=2B
    g1[1] = (tensor_d0 & 0xFFFFu) << 16;                     // tensor_dim0[15:0]
    g1[2] = (tensor_d0 >> 16) | ((tensor_d1 & 0xFFFFu) << 16);
    g1[3] = (tensor_d1 >> 16) | ((row_elems & 0xFFFFu) << 16);  // tile_dim0
    g1[4] = rows & 0xFFFFu;                                  // tile_dim1, tile_dim2=0
    g1[5] = (u32)(stride_elems & 0xFFFFFFFFull);             // dim0 stride lo
    g1[6] = (u32)((stride_elems >> 32) & 0xFFFFull);         // stride hi | dim1stride=0
    g1[7] = 0u;
    asm volatile("tensor_load_to_lds %0, %1" :: "s"(g0), "s"(g1) : "memory");
}

// ---------------- f32 -> bf16 bulk convert ----------------
__global__ void k_cvt_bf16(const float* __restrict__ src,
                           unsigned short* __restrict__ dst, int n) {
    int i = blockIdx.x * blockDim.x + threadIdx.x;
    int stride = gridDim.x * blockDim.x;
    for (; i < n; i += stride) dst[i] = f2bf(src[i]);
}

__global__ void k_fill_zero16(unsigned short* __restrict__ dst, int n) {
    int i = blockIdx.x * blockDim.x + threadIdx.x;
    if (i < n) dst[i] = 0;
}

// ---------------- WMMA GEMM: C[M, Nt*64] = A[M,K] * B[Nt*64, K]^T ------------
// bf16 in, f32 out. One wave -> 16x64 slab, branch-free (EXEC all-ones).
// A tile (16 x KCH) double-buffered in LDS via TDM. M%16==0, K%KCH==0.
__global__ void __launch_bounds__(256)
k_gemm_bf16(const unsigned short* __restrict__ A,
            const unsigned short* __restrict__ B,
            float* __restrict__ C, int M, int Ntiles64, int K, int ldc) {
    __shared__ __align__(1024) unsigned short smA[8][2][16 * KCH];
    const int lane = threadIdx.x & 31;
    const int wave = threadIdx.x >> 5;
    const int job  = blockIdx.x * 8 + wave;
    const int mt   = job / Ntiles64;
    const int nt   = job - mt * Ntiles64;
    if (mt >= (M >> 4)) return;
    const int m0 = mt << 4;
    const int n0 = nt << 6;
    const int r  = lane & 15;
    const int hi = lane >> 4;
    const u32 lds0 = (u32)(unsigned long long)(&smA[wave][0][0]);
    const unsigned short* Abase = A + (size_t)m0 * K;

    // per-column-tile B row base pointers (lane-fixed, K-invariant)
    const unsigned short* Brow[4];
    #pragma unroll
    for (int i = 0; i < 4; ++i)
        Brow[i] = B + (size_t)(n0 + (i << 4) + r) * K + (hi << 3);

    // prologue: DMA first A chunk
    tdm_load_tile_2d(lds0, Abase, KCH, 16, (u32)K, (u32)M, (unsigned long long)K);

    v8f acc[4] = {};
    for (int k0 = 0; k0 < K; k0 += KCH) {
        const int cur = (k0 / KCH) & 1;
        if (k0 + KCH < K) {
            // DMA next chunk into the other buffer, overlap with compute
            tdm_load_tile_2d(lds0 + (cur ^ 1) * TILE_BYTES, Abase + k0 + KCH,
                             KCH, 16, (u32)K, (u32)M, (unsigned long long)K);
            __builtin_prefetch(Brow[0] + k0 + KCH, 0, 1);
            __builtin_amdgcn_s_wait_tensorcnt(1);   // current chunk resident
        } else {
            __builtin_amdgcn_s_wait_tensorcnt(0);
        }
        const unsigned short* tile = &smA[wave][cur][0];

        // hoist all 4 A fragments of this chunk (one ds clause, 8x ds_load_b128)
        v16bf a[4];
        #pragma unroll
        for (int kk4 = 0; kk4 < 4; ++kk4) {
            const unsigned short* p = tile + r * KCH + (kk4 << 5) + (hi << 3);
            ((float4*)&a[kk4])[0] = *(const float4*)(p);
            ((float4*)&a[kk4])[1] = *(const float4*)(p + 16);
        }
        #pragma unroll
        for (int kk4 = 0; kk4 < 4; ++kk4) {
            // load all 4 B fragments first (8 global_load_b128 in flight),
            // then issue the 4 WMMAs — avoids load->wait->wmma serialization
            v16bf b[4];
            #pragma unroll
            for (int i = 0; i < 4; ++i) {
                const unsigned short* pb = Brow[i] + k0 + (kk4 << 5);
                ((float4*)&b[i])[0] = *(const float4*)(pb);
                ((float4*)&b[i])[1] = *(const float4*)(pb + 16);
            }
            #pragma unroll
            for (int i = 0; i < 4; ++i)
                acc[i] = __builtin_amdgcn_wmma_f32_16x16x32_bf16(
                    false, a[kk4], false, b[i], (short)0, acc[i], false, false);
        }
    }
    // C fragment: lane col n+r, rows m0 + j + 8*hi
    const int rb = m0 + (hi << 3);
    #pragma unroll
    for (int i = 0; i < 4; ++i) {
        int col = n0 + (i << 4) + r;
        #pragma unroll
        for (int j = 0; j < 8; ++j)
            C[(size_t)(rb + j) * ldc + col] = acc[i][j];
    }
}

// ---------------- depthwise causal conv (K=4) + bias + SiLU ----------------
__global__ void k_conv_silu(const float* __restrict__ xz,
                            const float* __restrict__ cw,
                            const float* __restrict__ cb,
                            float* __restrict__ hconv_f,
                            unsigned short* __restrict__ hconv_b) {
    int idx = blockIdx.x * blockDim.x + threadIdx.x;
    if (idx >= TSEQ * D_INNER) return;
    int t = idx >> 11;
    int e = idx & (D_INNER - 1);
    float acc = cb[e];
    #pragma unroll
    for (int j = 0; j < D_CONV; ++j) {
        int ts = t + j - (D_CONV - 1);
        if (ts >= 0)
            acc += xz[(size_t)ts * (2 * D_INNER) + e] * cw[e * D_CONV + j];
    }
    float h = acc * sigm(acc);
    hconv_f[idx] = h;
    hconv_b[idx] = f2bf(h);
}

// ---------------- chunked selective scan ----------------
// pass1: per (e, chunk) compute product of A_bar and chunk-final state
__global__ void k_scan_pass1(const float* __restrict__ dBC,
                             const float* __restrict__ hconv,
                             const float* __restrict__ A_log,
                             float* __restrict__ chunkAcc) {
    int tid = blockIdx.x * blockDim.x + threadIdx.x;
    if (tid >= D_INNER * NCHUNK) return;
    int e = tid & (D_INNER - 1);
    int c = tid >> 11;
    float aneg[D_STATE], Ac[D_STATE], h[D_STATE];
    #pragma unroll
    for (int n = 0; n < D_STATE; ++n) {
        aneg[n] = -__expf(A_log[e * D_STATE + n]);
        Ac[n] = 1.f; h[n] = 0.f;
    }
    int t0 = c * CLEN;
    for (int t = t0; t < t0 + CLEN; ++t) {
        const float* row = dBC + (size_t)t * NPROJ_PAD;
        float d  = softp(row[e]);
        float du = d * hconv[(size_t)t * D_INNER + e];
        #pragma unroll
        for (int n = 0; n < D_STATE; ++n) {
            float ab = __expf(d * aneg[n]);
            h[n]  = ab * h[n] + du * row[D_INNER + n];
            Ac[n] *= ab;
        }
    }
    float* o = chunkAcc + (size_t)tid * (2 * D_STATE);
    #pragma unroll
    for (int n = 0; n < D_STATE; ++n) { o[n] = Ac[n]; o[D_STATE + n] = h[n]; }
}

// pass2: exclusive combine across the 32 chunks per (e, n)
__global__ void k_scan_pass2(const float* __restrict__ chunkAcc,
                             float* __restrict__ hinit) {
    int tid = blockIdx.x * blockDim.x + threadIdx.x;
    if (tid >= D_INNER * D_STATE) return;
    int n = tid & (D_STATE - 1);
    int e = tid >> 4;
    float run = 0.f;
    for (int c = 0; c < NCHUNK; ++c) {
        size_t base = ((size_t)c * D_INNER + e) * (2 * D_STATE);
        hinit[((size_t)c * D_INNER + e) * D_STATE + n] = run;
        run = chunkAcc[base + n] * run + chunkAcc[base + D_STATE + n];
    }
}

// pass3: re-scan each chunk with carried-in state; fused C.h + D.u, silu(z), bf16 cast
__global__ void k_scan_pass3(const float* __restrict__ dBC,
                             const float* __restrict__ hconv,
                             const float* __restrict__ A_log,
                             const float* __restrict__ Dvec,
                             const float* __restrict__ xz,
                             const float* __restrict__ hinit,
                             unsigned short* __restrict__ y_bf) {
    int tid = blockIdx.x * blockDim.x + threadIdx.x;
    if (tid >= D_INNER * NCHUNK) return;
    int e = tid & (D_INNER - 1);
    int c = tid >> 11;
    float aneg[D_STATE], h[D_STATE];
    #pragma unroll
    for (int n = 0; n < D_STATE; ++n) {
        aneg[n] = -__expf(A_log[e * D_STATE + n]);
        h[n] = hinit[(size_t)tid * D_STATE + n];
    }
    float Dv = Dvec[e];
    int t0 = c * CLEN;
    for (int t = t0; t < t0 + CLEN; ++t) {
        const float* row = dBC + (size_t)t * NPROJ_PAD;
        float u  = hconv[(size_t)t * D_INNER + e];
        float d  = softp(row[e]);
        float du = d * u;
        float y = 0.f;
        #pragma unroll
        for (int n = 0; n < D_STATE; ++n) {
            float ab = __expf(d * aneg[n]);
            h[n] = ab * h[n] + du * row[D_INNER + n];
            y += h[n] * row[D_INNER + D_STATE + n];
        }
        y += u * Dv;
        float z = xz[(size_t)t * (2 * D_INNER) + D_INNER + e];
        y *= z * sigm(z);
        y_bf[(size_t)t * D_INNER + e] = f2bf(y);
    }
}

// ---------------- launcher ----------------
extern "C" void kernel_launch(void* const* d_in, const int* in_sizes, int n_in,
                              void* d_out, int out_size, void* d_ws, size_t ws_size,
                              hipStream_t stream) {
    (void)in_sizes; (void)n_in; (void)out_size; (void)ws_size;
    const float* x     = (const float*)d_in[0];
    const float* W_in  = (const float*)d_in[1];
    const float* cw    = (const float*)d_in[2];
    const float* cb    = (const float*)d_in[3];
    const float* W_x   = (const float*)d_in[4];
    const float* A_log = (const float*)d_in[5];
    const float* Dv    = (const float*)d_in[6];
    const float* W_out = (const float*)d_in[7];
    float* out = (float*)d_out;

    char* ws = (char*)d_ws;
    size_t off = 0;
    auto carve = [&](size_t bytes) -> char* {
        char* p = ws + off;
        off += (bytes + 255) & ~(size_t)255;
        return p;
    };
    unsigned short* Win_b   = (unsigned short*)carve((size_t)2 * D_INNER * D_MODEL * 2);
    unsigned short* Wx_b    = (unsigned short*)carve((size_t)NPROJ_PAD * D_INNER * 2); // 32 pad rows
    unsigned short* Wout_b  = (unsigned short*)carve((size_t)D_MODEL * D_INNER * 2);
    unsigned short* x_b     = (unsigned short*)carve((size_t)TSEQ * D_MODEL * 2);
    float*          xz      = (float*)carve((size_t)TSEQ * 2 * D_INNER * 4);
    float*          hconv_f = (float*)carve((size_t)TSEQ * D_INNER * 4);
    unsigned short* hconv_b = (unsigned short*)carve((size_t)TSEQ * D_INNER * 2);
    float*          dBC     = (float*)carve((size_t)TSEQ * NPROJ_PAD * 4);
    float*          cAcc    = (float*)carve((size_t)D_INNER * NCHUNK * 2 * D_STATE * 4);
    float*          hinit   = (float*)carve((size_t)D_INNER * NCHUNK * D_STATE * 4);
    unsigned short* y_b     = (unsigned short*)carve((size_t)TSEQ * D_INNER * 2);

    // bf16 conversions (weights + activations); zero the 32 W_x pad rows
    k_cvt_bf16<<<2048, 256, 0, stream>>>(W_in,  Win_b,  2 * D_INNER * D_MODEL);
    k_cvt_bf16<<<2048, 256, 0, stream>>>(W_x,   Wx_b,   NPROJ * D_INNER);
    k_fill_zero16<<<((NPROJ_PAD - NPROJ) * D_INNER + 255) / 256, 256, 0, stream>>>(
        Wx_b + (size_t)NPROJ * D_INNER, (NPROJ_PAD - NPROJ) * D_INNER);
    k_cvt_bf16<<<2048, 256, 0, stream>>>(W_out, Wout_b, D_MODEL * D_INNER);
    k_cvt_bf16<<<2048, 256, 0, stream>>>(x,     x_b,    TSEQ * D_MODEL);

    // xz = x @ W_in^T   (2048 x 4096, K=1024)
    {
        int nt = (2 * D_INNER) / 64, jobs = (TSEQ / 16) * nt;
        k_gemm_bf16<<<(jobs + 7) / 8, 256, 0, stream>>>(x_b, Win_b, xz,
                                                        TSEQ, nt, D_MODEL, 2 * D_INNER);
    }
    // depthwise conv + SiLU
    k_conv_silu<<<(TSEQ * D_INNER + 255) / 256, 256, 0, stream>>>(xz, cw, cb,
                                                                  hconv_f, hconv_b);
    // dBC = h_conv @ W_x^T (2048 x 2112(pad), K=2048)
    {
        int nt = NPROJ_PAD / 64, jobs = (TSEQ / 16) * nt;
        k_gemm_bf16<<<(jobs + 7) / 8, 256, 0, stream>>>(hconv_b, Wx_b, dBC,
                                                        TSEQ, nt, D_INNER, NPROJ_PAD);
    }
    // chunked selective scan + epilogue
    k_scan_pass1<<<(D_INNER * NCHUNK + 255) / 256, 256, 0, stream>>>(dBC, hconv_f,
                                                                     A_log, cAcc);
    k_scan_pass2<<<(D_INNER * D_STATE + 255) / 256, 256, 0, stream>>>(cAcc, hinit);
    k_scan_pass3<<<(D_INNER * NCHUNK + 255) / 256, 256, 0, stream>>>(dBC, hconv_f,
                                                                     A_log, Dv, xz,
                                                                     hinit, y_b);
    // out = y @ W_out^T (2048 x 1024, K=2048)
    {
        int nt = D_MODEL / 64, jobs = (TSEQ / 16) * nt;
        k_gemm_bf16<<<(jobs + 7) / 8, 256, 0, stream>>>(y_b, Wout_b, out,
                                                        TSEQ, nt, D_INNER, D_MODEL);
    }
}
